// EvoformerBlock_10849087390541
// MI455X (gfx1250) — matrix operand order repaired and checked by reading, not compiled, not run on hardware
//
#include <hip/hip_runtime.h>
#include <hip/hip_bf16.h>

// ---------------------------------------------------------------------------
// Types & helpers for CDNA5 (gfx1250) WMMA, wave32
// ---------------------------------------------------------------------------
typedef __bf16 v16bf __attribute__((ext_vector_type(16)));
typedef float  v8f   __attribute__((ext_vector_type(8)));
typedef unsigned int u32x4 __attribute__((ext_vector_type(4)));
typedef unsigned int u32x8 __attribute__((ext_vector_type(8)));

union BF16x16 { unsigned int u[8]; v16bf v; };

__device__ __forceinline__ unsigned short f2bf(float f) {
  unsigned int u = __float_as_uint(f);
  u += 0x7FFFu + ((u >> 16) & 1u);          // round-to-nearest-even
  return (unsigned short)(u >> 16);
}

// A-operand (16x32 bf16) k index for VGPR v, lane-half h  (ISA 7.12.2)
__device__ __forceinline__ int kmapA(int v, int h) {
  return (v < 4 ? 2 * v : 16 + 2 * (v - 4)) + 8 * h;
}
// B-operand (32x16 bf16) k index for VGPR v, lane-half h
__device__ __forceinline__ int kmapB(int v, int h) {
  return (v < 4 ? 2 * v : 16 + 2 * (v - 4)) + 8 * h;
}

#define WMMA_BF16(A, B, C) \
  __builtin_amdgcn_wmma_f32_16x16x32_bf16(false, (A), false, (B), (short)0, (C), false, false)

// ---------------------------------------------------------------------------
// Tensor Data Mover: stage a contiguous bf16 panel (64 rows x K) into LDS.
// 1-D tile, data_size=2B.  LDS padding: 1 DWORD inserted every row-length
// interval (K/2 dwords, K power of two) -> LDS row stride = K+2 halfwords,
// odd dword stride -> bank-conflict-free column reads.
// Issued by one wave; caller waits TENSORcnt then barriers.
// ---------------------------------------------------------------------------
__device__ __forceinline__ void tdm_load_panel(const unsigned short* src,
                                               unsigned int ldsOff, int K) {
  unsigned long long ga = (unsigned long long)src;
  unsigned int nk = 64u * (unsigned int)K;                 // elements (<= 32768)
  unsigned int padc = (unsigned int)(31 - __builtin_clz((unsigned)K)) - 2u; // K=128->5,256->6,512->7
  u32x4 g0;
  g0[0] = 1u;                                              // count=1, user mode
  g0[1] = ldsOff;                                          // lds_addr
  g0[2] = (unsigned int)ga;                                // global_addr[31:0]
  g0[3] = (unsigned int)(ga >> 32) | (2u << 30);           // global_addr[56:32] | type=2
  u32x8 g1;
  g1[0] = (1u << 16) | (1u << 20) | (padc << 22);          // data_size=2B | pad_en | pad_interval
  g1[1] = (nk & 0xFFFFu) << 16;                            // tensor_dim0 lo16
  g1[2] = (nk >> 16) | (1u << 16);                         // tensor_dim0 hi | tensor_dim1=1
  g1[3] = (nk & 0xFFFFu) << 16;                            // tile_dim0
  g1[4] = 0u;                                              // tile_dim1/2 unused
  g1[5] = nk;                                              // tensor_dim0_stride lo32
  g1[6] = 0u;
  g1[7] = 0u;
  asm volatile("tensor_load_to_lds %0, %1" :: "s"(g0), "s"(g1) : "memory");
}

// ---------------------------------------------------------------------------
// Weight convert: f32 (K,N) row-major  ->  bf16 (N,K) row-major (transposed)
// ---------------------------------------------------------------------------
__global__ void cvtw_kernel(const float* __restrict__ w, unsigned short* __restrict__ wt,
                            int K, int N) {
  int i = blockIdx.x * 256 + threadIdx.x;
  if (i >= K * N) return;
  int n = i / K, k = i - n * K;
  wt[i] = f2bf(w[(size_t)k * N + n]);
}

// ---------------------------------------------------------------------------
// LayerNorm over last dim 128, f32 in -> bf16 out.  One wave per row.
// ---------------------------------------------------------------------------
__global__ void ln_kernel(const float* __restrict__ x, const float* __restrict__ g,
                          const float* __restrict__ b, unsigned short* __restrict__ y,
                          int rows) {
  int wave = threadIdx.x >> 5, lane = threadIdx.x & 31;
  int row = blockIdx.x * (blockDim.x >> 5) + wave;
  if (row >= rows) return;
  const float4 v = *(const float4*)(x + (size_t)row * 128 + lane * 4);
  float s = v.x + v.y + v.z + v.w;
  for (int o = 16; o; o >>= 1) s += __shfl_xor(s, o, 32);
  float mu = s * (1.f / 128.f);
  float dx = v.x - mu, dy = v.y - mu, dz = v.z - mu, dw = v.w - mu;
  float q = dx * dx + dy * dy + dz * dz + dw * dw;
  for (int o = 16; o; o >>= 1) q += __shfl_xor(q, o, 32);
  float inv = rsqrtf(q * (1.f / 128.f) + 1e-5f);
  int c = lane * 4;
  unsigned int p0 = (unsigned int)f2bf(dx * inv * g[c + 0] + b[c + 0]) |
                    ((unsigned int)f2bf(dy * inv * g[c + 1] + b[c + 1]) << 16);
  unsigned int p1 = (unsigned int)f2bf(dz * inv * g[c + 2] + b[c + 2]) |
                    ((unsigned int)f2bf(dw * inv * g[c + 3] + b[c + 3]) << 16);
  uint2 o2; o2.x = p0; o2.y = p1;
  *(uint2*)(y + (size_t)row * 128 + c) = o2;
}

// ---------------------------------------------------------------------------
// Generic GEMM:  C(M,N) = A(M,K)bf16 @ Wt(N,K)bf16^T + bias (+gelu) (+res)
// block = 128 thr (4 waves), block tile 64x64, wave tile 32x32 (2x2 WMMA).
// B panel (64 x K) staged to LDS via TDM; A ping-pong double-buffered from
// global (loop unrolled by 2 k-steps, K multiple of 64 -> no register copies,
// no WMMA->VALU hazard nops).  grid(M/64, N/64); dyn LDS = 64*(K+2)*2 B.
// ---------------------------------------------------------------------------
__global__ void gemm_kernel(const unsigned short* __restrict__ A,
                            const unsigned short* __restrict__ Bt,
                            const float* __restrict__ bias,
                            const float* __restrict__ res,
                            float* __restrict__ outF,
                            unsigned short* __restrict__ outBf,
                            int M, int N, int K, int gelu) {
  extern __shared__ unsigned short sB[];
  int lane = threadIdx.x & 31, w = threadIdx.x >> 5;
  int half = lane >> 4, lp = lane & 15;
  int m0 = blockIdx.x * 64, n0 = blockIdx.y * 64;
  if (w == 0) {
    tdm_load_panel(Bt + (size_t)n0 * K, (unsigned int)(unsigned long long)sB, K);
    __builtin_amdgcn_s_wait_tensorcnt(0);
  }
  __syncthreads();

  int KS = K + 2;                                   // padded LDS row stride
  int mw = m0 + (w & 1) * 32;
  int nw = (w >> 1) * 32;
  const unsigned short* ar[2] = { A + (size_t)(mw + lp) * K,
                                  A + (size_t)(mw + 16 + lp) * K };
  const unsigned short* br[2] = { sB + (size_t)(nw + lp) * KS,
                                  sB + (size_t)(nw + 16 + lp) * KS };
  v8f c[2][2];
#pragma unroll
  for (int ni = 0; ni < 2; ++ni) {
    float bb = bias ? bias[n0 + nw + ni * 16 + lp] : 0.f;
#pragma unroll
    for (int mi = 0; mi < 2; ++mi) c[mi][ni] = (v8f){bb, bb, bb, bb, bb, bb, bb, bb};
  }

  auto loadA = [&](BF16x16* dst, int kk) {
#pragma unroll
    for (int mi = 0; mi < 2; ++mi)
#pragma unroll
      for (int v = 0; v < 8; ++v)
        dst[mi].u[v] = *(const unsigned int*)(ar[mi] + kk + kmapA(v, half));
  };
  auto step = [&](BF16x16* a, int kk) {
    BF16x16 b0, b1;
#pragma unroll
    for (int v = 0; v < 8; ++v) {
      b0.u[v] = *(const unsigned int*)(br[0] + kk + kmapB(v, half));
      b1.u[v] = *(const unsigned int*)(br[1] + kk + kmapB(v, half));
    }
    c[0][0] = WMMA_BF16(a[0].v, b0.v, c[0][0]);
    c[0][1] = WMMA_BF16(a[0].v, b1.v, c[0][1]);
    c[1][0] = WMMA_BF16(a[1].v, b0.v, c[1][0]);
    c[1][1] = WMMA_BF16(a[1].v, b1.v, c[1][1]);
  };

  BF16x16 aP0[2], aP1[2];
  loadA(aP0, 0);
  for (int k0 = 0; k0 < K; k0 += 64) {           // K is a multiple of 64
    loadA(aP1, k0 + 32);
    __builtin_prefetch(ar[0] + k0 + 64, 0, 1);
    step(aP0, k0);
    if (k0 + 64 < K) loadA(aP0, k0 + 64);
    step(aP1, k0 + 32);
  }
#pragma unroll
  for (int mi = 0; mi < 2; ++mi)
#pragma unroll
    for (int ni = 0; ni < 2; ++ni)
#pragma unroll
      for (int r = 0; r < 8; ++r) {
        int mm = mw + mi * 16 + r + 8 * half;
        int nn = n0 + nw + ni * 16 + lp;
        float val = c[mi][ni][r];
        if (gelu) val = 0.5f * val * (1.f + erff(val * 0.70710678118654752f));
        if (res)   val += res[(size_t)mm * N + nn];
        if (outF)  outF[(size_t)mm * N + nn] = val;
        if (outBf) outBf[(size_t)mm * N + nn] = f2bf(val);
      }
}

// ---------------------------------------------------------------------------
// Triangle batched GEMM: for each d (grid.z): X_d(L,L) = Ad(L,L) @ Bd(L,L)
// (B accessed as N x K rows). Output scattered to (i, n, d) f32 so the
// following LayerNorm over d is contiguous.  Same blocking/TDM/pipeline.
// ---------------------------------------------------------------------------
__global__ void gemm_tri_kernel(const unsigned short* __restrict__ A,
                                const unsigned short* __restrict__ Bt,
                                float* __restrict__ outX, int L) {
  extern __shared__ unsigned short sB[];
  int lane = threadIdx.x & 31, w = threadIdx.x >> 5;
  int half = lane >> 4, lp = lane & 15;
  int d = blockIdx.z;
  const unsigned short* Ad = A  + (size_t)d * L * L;
  const unsigned short* Bd = Bt + (size_t)d * L * L;
  int m0 = blockIdx.x * 64, n0 = blockIdx.y * 64;
  if (w == 0) {
    tdm_load_panel(Bd + (size_t)n0 * L, (unsigned int)(unsigned long long)sB, L);
    __builtin_amdgcn_s_wait_tensorcnt(0);
  }
  __syncthreads();

  int KS = L + 2;
  int mw = m0 + (w & 1) * 32;
  int nw = (w >> 1) * 32;
  const unsigned short* ar[2] = { Ad + (size_t)(mw + lp) * L,
                                  Ad + (size_t)(mw + 16 + lp) * L };
  const unsigned short* br[2] = { sB + (size_t)(nw + lp) * KS,
                                  sB + (size_t)(nw + 16 + lp) * KS };
  v8f c[2][2];
#pragma unroll
  for (int mi = 0; mi < 2; ++mi)
#pragma unroll
    for (int ni = 0; ni < 2; ++ni) c[mi][ni] = (v8f){0.f, 0.f, 0.f, 0.f, 0.f, 0.f, 0.f, 0.f};

  auto loadA = [&](BF16x16* dst, int kk) {
#pragma unroll
    for (int mi = 0; mi < 2; ++mi)
#pragma unroll
      for (int v = 0; v < 8; ++v)
        dst[mi].u[v] = *(const unsigned int*)(ar[mi] + kk + kmapA(v, half));
  };
  auto step = [&](BF16x16* a, int kk) {
    BF16x16 b0, b1;
#pragma unroll
    for (int v = 0; v < 8; ++v) {
      b0.u[v] = *(const unsigned int*)(br[0] + kk + kmapB(v, half));
      b1.u[v] = *(const unsigned int*)(br[1] + kk + kmapB(v, half));
    }
    c[0][0] = WMMA_BF16(a[0].v, b0.v, c[0][0]);
    c[0][1] = WMMA_BF16(a[0].v, b1.v, c[0][1]);
    c[1][0] = WMMA_BF16(a[1].v, b0.v, c[1][0]);
    c[1][1] = WMMA_BF16(a[1].v, b1.v, c[1][1]);
  };

  BF16x16 aP0[2], aP1[2];
  loadA(aP0, 0);
  for (int k0 = 0; k0 < L; k0 += 64) {           // L = 256, multiple of 64
    loadA(aP1, k0 + 32);
    __builtin_prefetch(ar[0] + k0 + 64, 0, 1);
    step(aP0, k0);
    if (k0 + 64 < L) loadA(aP0, k0 + 64);
    step(aP1, k0 + 32);
  }
#pragma unroll
  for (int mi = 0; mi < 2; ++mi)
#pragma unroll
    for (int ni = 0; ni < 2; ++ni)
#pragma unroll
      for (int r = 0; r < 8; ++r) {
        int mm = mw + mi * 16 + r + 8 * half;
        int nn = n0 + nw + ni * 16 + lp;
        outX[((size_t)mm * L + nn) * 128 + d] = c[mi][ni][r];
      }
}

// ---------------------------------------------------------------------------
// Fused attention, one wave per (q-tile16, head, row):
//   logits = Q K^T / sqrt(32) (+bias)  ->  softmax  ->  P V  -> bf16 out
// qkv layout: (OUTER*L, 384) bf16, cols = [q | k | v], head h = 32 cols each.
// ---------------------------------------------------------------------------
__global__ void attn_kernel(const unsigned short* __restrict__ qkv,
                            const float* __restrict__ bias,   // (4, L, L) or null
                            unsigned short* __restrict__ out, // (OUTER*L, 128)
                            int L) {
  __shared__ float sm[16][264];
  int lane = threadIdx.x & 31;
  int half = lane >> 4, lp = lane & 15;
  int qt = blockIdx.x, h = blockIdx.y, z = blockIdx.z;
  size_t rowbase = (size_t)z * L;

  BF16x16 aq;
  {
    const unsigned short* qrow = qkv + (rowbase + qt * 16 + lp) * 384 + h * 32;
#pragma unroll
    for (int v = 0; v < 8; ++v) aq.u[v] = *(const unsigned int*)(qrow + kmapA(v, half));
  }
  for (int jt = 0; jt < (L >> 4); ++jt) {
    BF16x16 bk;
    const unsigned short* krow = qkv + (rowbase + jt * 16 + lp) * 384 + 128 + h * 32;
#pragma unroll
    for (int v = 0; v < 8; ++v) bk.u[v] = *(const unsigned int*)(krow + kmapB(v, half));
    v8f cc = {0.f, 0.f, 0.f, 0.f, 0.f, 0.f, 0.f, 0.f};
    cc = WMMA_BF16(aq.v, bk.v, cc);
#pragma unroll
    for (int r = 0; r < 8; ++r) {
      int m = r + 8 * half, j = jt * 16 + lp;
      float val = cc[r] * 0.17677669529663688f;   // 1/sqrt(32)
      if (bias) val += bias[((size_t)h * L + qt * 16 + m) * L + j];
      sm[m][j] = val;
    }
  }
  __syncthreads();
  {
    int c0 = half * (L >> 1);
    float mx = -1e30f;
    for (int j = 0; j < (L >> 1); ++j) mx = fmaxf(mx, sm[lp][c0 + j]);
    mx = fmaxf(mx, __shfl_xor(mx, 16, 32));
    float s = 0.f;
    for (int j = 0; j < (L >> 1); ++j) {
      float e = __expf(sm[lp][c0 + j] - mx);
      sm[lp][c0 + j] = e; s += e;
    }
    s += __shfl_xor(s, 16, 32);
    float inv = 1.f / s;
    for (int j = 0; j < (L >> 1); ++j) sm[lp][c0 + j] *= inv;
  }
  __syncthreads();
  v8f c0a = {0.f, 0.f, 0.f, 0.f, 0.f, 0.f, 0.f, 0.f};
  v8f c1a = {0.f, 0.f, 0.f, 0.f, 0.f, 0.f, 0.f, 0.f};
  for (int kt = 0; kt < (L >> 5); ++kt) {
    BF16x16 ap, bv0, bv1;
#pragma unroll
    for (int v = 0; v < 8; ++v) {
      int kk = kt * 32 + kmapA(v, half);
      unsigned int lo = f2bf(sm[lp][kk]);
      unsigned int hi = f2bf(sm[lp][kk + 1]);
      ap.u[v] = lo | (hi << 16);
    }
#pragma unroll
    for (int v = 0; v < 8; ++v) {
      int kk = kt * 32 + kmapB(v, half);
      const unsigned short* vr = qkv + (rowbase + kk) * 384 + 256 + h * 32;
      unsigned int e0 = vr[lp],      e1 = vr[384 + lp];
      unsigned int f0 = vr[16 + lp], f1 = vr[384 + 16 + lp];
      bv0.u[v] = e0 | (e1 << 16);
      bv1.u[v] = f0 | (f1 << 16);
    }
    c0a = WMMA_BF16(ap.v, bv0.v, c0a);
    c1a = WMMA_BF16(ap.v, bv1.v, c1a);
  }
#pragma unroll
  for (int r = 0; r < 8; ++r) {
    int m = r + 8 * half;
    unsigned short* orow = out + (rowbase + qt * 16 + m) * 128 + h * 32;
    orow[lp]      = f2bf(c0a[r]);
    orow[16 + lp] = f2bf(c1a[r]);
  }
}

// ---------------------------------------------------------------------------
// Elementwise kernels
// ---------------------------------------------------------------------------
__global__ void mulsig_kernel(const float* __restrict__ p, const float* __restrict__ g,
                              unsigned short* __restrict__ o, int n) {
  int i = blockIdx.x * 256 + threadIdx.x;
  if (i >= n) return;
  float gg = g[i];
  o[i] = f2bf(p[i] * (1.f / (1.f + __expf(-gg))));
}

__global__ void gateres_kernel(const float* __restrict__ z, const float* __restrict__ po,
                               const float* __restrict__ go, float* __restrict__ o, int n) {
  int i = blockIdx.x * 256 + threadIdx.x;
  if (i >= n) return;
  float gg = go[i];
  o[i] = z[i] + po[i] * (1.f / (1.f + __expf(-gg)));
}

// bf16 (L,L,128) -> (128,L,L); swap=1 transposes the two spatial dims
__global__ void trT_kernel(const unsigned short* __restrict__ in,
                           unsigned short* __restrict__ o, int swap) {
  int i = blockIdx.x * 256 + threadIdx.x;     // 128*256*256 total
  int d = i >> 16, rem = i & 0xFFFF;
  int x = rem >> 8, y = rem & 255;
  int src = swap ? (y * 256 + x) : (x * 256 + y);
  o[i] = in[(size_t)src * 128 + d];
}

// f32 (L,L,128) spatial transpose
__global__ void trpair_kernel(const float* __restrict__ in, float* __restrict__ o) {
  int i = blockIdx.x * 256 + threadIdx.x;     // 256*256*128 total
  int d = i & 127, xy = i >> 7;
  int x = xy >> 8, y = xy & 255;
  o[i] = in[((size_t)(y * 256 + x)) * 128 + d];
}

// bias[h][x][y] = pair[x][y][:] . W[:,h] + b[h]
__global__ void bias_kernel(const float* __restrict__ pair, const float* __restrict__ w,
                            const float* __restrict__ b, float* __restrict__ out) {
  int i = blockIdx.x * 256 + threadIdx.x;     // 4*65536
  int h = i >> 16, xy = i & 0xFFFF;
  const float* row = pair + (size_t)xy * 128;
  float s = b[h];
  for (int c = 0; c < 128; ++c) s += row[c] * w[c * 4 + h];
  out[(size_t)h * 65536 + xy] = s;
}

// ---------------------------------------------------------------------------
// Host orchestration
// ---------------------------------------------------------------------------
enum {
  IN_MSA = 0, IN_PAIR = 1, P_BIAS_W = 2, P_BIAS_B = 3,
  MA = 4,   // msa_attn: +0 ng,+1 nb,+2 qkvw,+3 qkvb,+4 pw,+5 pb
  MF = 10,  // msa_ffn : +0 ng,+1 nb,+2 w1w,+3 w1b,+4 w2w,+5 w2b
  TO = 16, TI = 32,  // tri: +0 ng,+1 nb,+2 paw,+3 pab,+4 pbw,+5 pbb,+6 gaw,+7 gab,
                     //      +8 gbw,+9 gbb,+10 pow,+11 pob,+12 gow,+13 gob,+14 nog,+15 nob
  PR = 48, PC = 54,  // pair attn: same layout as MA
  PF = 60            // pair ffn: same layout as MF
};

extern "C" void kernel_launch(void* const* d_in, const int* in_sizes, int n_in,
                              void* d_out, int out_size, void* d_ws, size_t ws_size,
                              hipStream_t stream) {
  (void)in_sizes; (void)n_in; (void)out_size; (void)ws_size;
  auto P = [&](int i) { return (const float*)d_in[i]; };
  const float* msa_in  = P(IN_MSA);
  const float* pair_in = P(IN_PAIR);
  float* outMsa  = (float*)d_out;
  float* outPair = outMsa + (size_t)64 * 256 * 128;

  const int RM = 64 * 256;        // msa rows
  const int RP = 256 * 256;       // pair rows
  const int L  = 256;

  char* ws = (char*)d_ws;
  size_t off = 0;
  auto alloc = [&](size_t bytes) {
    off = (off + 255) & ~(size_t)255;
    void* p = ws + off; off += bytes; return p;
  };

  auto makeWt = [&](int idx, int K, int N) {
    unsigned short* wt = (unsigned short*)alloc((size_t)K * N * 2);
    cvtw_kernel<<<(K * N + 255) / 256, 256, 0, stream>>>(P(idx), wt, K, N);
    return wt;
  };
  unsigned short* wt_mqkv  = makeWt(MA + 2, 128, 384);
  unsigned short* wt_mproj = makeWt(MA + 4, 128, 128);
  unsigned short* wt_mw1   = makeWt(MF + 2, 128, 512);
  unsigned short* wt_mw2   = makeWt(MF + 4, 512, 128);
  unsigned short* wt_to[6], *wt_ti[6];
  for (int j = 0; j < 6; ++j) wt_to[j] = makeWt(TO + 2 + 2 * j, 128, 128); // pa,pb,ga,gb,po,go
  for (int j = 0; j < 6; ++j) wt_ti[j] = makeWt(TI + 2 + 2 * j, 128, 128);
  unsigned short* wt_prqkv  = makeWt(PR + 2, 128, 384);
  unsigned short* wt_prproj = makeWt(PR + 4, 128, 128);
  unsigned short* wt_pcqkv  = makeWt(PC + 2, 128, 384);
  unsigned short* wt_pcproj = makeWt(PC + 4, 128, 128);
  unsigned short* wt_fw1    = makeWt(PF + 2, 128, 512);
  unsigned short* wt_fw2    = makeWt(PF + 4, 512, 128);

  float* biasbuf = (float*)alloc((size_t)4 * RP * 4);
  float* pairA   = (float*)alloc((size_t)RP * 128 * 4);
  float* pairB   = (float*)alloc((size_t)RP * 128 * 4);
  float* f32A    = (float*)alloc((size_t)RP * 128 * 4);
  float* f32B    = (float*)alloc((size_t)RP * 128 * 4);
  float* f32C    = (float*)alloc((size_t)RP * 128 * 4);
  unsigned short* bfzn = (unsigned short*)alloc((size_t)RP * 128 * 2);
  unsigned short* bfA  = (unsigned short*)alloc((size_t)RP * 128 * 2);
  unsigned short* bfB  = (unsigned short*)alloc((size_t)RP * 128 * 2);
  unsigned short* bfC  = (unsigned short*)alloc((size_t)RP * 128 * 2);
  unsigned short* bfD  = (unsigned short*)alloc((size_t)RP * 128 * 2);
  unsigned short* bigbf = (unsigned short*)alloc((size_t)RP * 512 * 2); // qkv / ffn hidden

  auto gemm = [&](const unsigned short* A, const unsigned short* Bt, const float* bias,
                  const float* res, float* oF, unsigned short* oB, int M, int N, int K,
                  int gelu) {
    size_t shmem = (size_t)64 * (K + 2) * 2;
    gemm_kernel<<<dim3(M / 64, N / 64), 128, shmem, stream>>>(A, Bt, bias, res, oF, oB,
                                                              M, N, K, gelu);
  };
  auto ln = [&](const float* x, const float* g, const float* b, unsigned short* y, int rows) {
    ln_kernel<<<rows / 8, 256, 0, stream>>>(x, g, b, y, rows);
  };

  // ===== bias = transpose(pair @ Wb + bb) =====
  bias_kernel<<<(4 * RP) / 256, 256, 0, stream>>>(pair_in, P(P_BIAS_W), P(P_BIAS_B), biasbuf);

  // ===== MSA row attention + FFN =====
  ln(msa_in, P(MA), P(MA + 1), bfzn, RM);
  gemm(bfzn, wt_mqkv, P(MA + 3), nullptr, nullptr, bigbf, RM, 384, 128, 0);
  attn_kernel<<<dim3(16, 4, 64), 32, 0, stream>>>(bigbf, biasbuf, bfA, L);
  gemm(bfA, wt_mproj, P(MA + 5), msa_in, f32A, nullptr, RM, 128, 128, 0);   // msa1
  ln(f32A, P(MF), P(MF + 1), bfzn, RM);
  gemm(bfzn, wt_mw1, P(MF + 3), nullptr, nullptr, bigbf, RM, 512, 128, 1);  // gelu -> bf16
  gemm(bigbf, wt_mw2, P(MF + 5), f32A, outMsa, nullptr, RM, 128, 512, 0);   // final msa

  // ===== triangle multiplication (shared routine) =====
  auto tri = [&](const float* zin, float* zout, int bp, unsigned short** wt, int outgoing) {
    ln(zin, P(bp), P(bp + 1), bfzn, RP);
    gemm(bfzn, wt[0], P(bp + 3), nullptr, f32A, nullptr, RP, 128, 128, 0);  // pa
    gemm(bfzn, wt[2], P(bp + 7), nullptr, f32B, nullptr, RP, 128, 128, 0);  // ga
    mulsig_kernel<<<(RP * 128) / 256, 256, 0, stream>>>(f32A, f32B, bfA, RP * 128);
    gemm(bfzn, wt[1], P(bp + 5), nullptr, f32A, nullptr, RP, 128, 128, 0);  // pb
    gemm(bfzn, wt[3], P(bp + 9), nullptr, f32B, nullptr, RP, 128, 128, 0);  // gb
    mulsig_kernel<<<(RP * 128) / 256, 256, 0, stream>>>(f32A, f32B, bfB, RP * 128);
    trT_kernel<<<(128 * RP) / 256, 256, 0, stream>>>(bfA, bfC, 0);          // aT[d][i][k]
    trT_kernel<<<(128 * RP) / 256, 256, 0, stream>>>(bfB, bfD, outgoing);   // bX
    gemm_tri_kernel<<<dim3(4, 4, 128), 128, (size_t)64 * (L + 2) * 2, stream>>>(bfC, bfD,
                                                                                f32A, L);
    ln(f32A, P(bp + 14), P(bp + 15), bfA, RP);                              // norm_o
    gemm(bfA, wt[4], P(bp + 11), nullptr, f32B, nullptr, RP, 128, 128, 0);  // po
    gemm(bfzn, wt[5], P(bp + 13), nullptr, f32C, nullptr, RP, 128, 128, 0); // go
    gateres_kernel<<<(RP * 128) / 256, 256, 0, stream>>>(zin, f32B, f32C, zout, RP * 128);
  };
  tri(pair_in, pairA, TO, wt_to, 1);   // outgoing
  tri(pairA,  pairB, TI, wt_ti, 0);    // incoming

  // ===== pair row / column attention =====
  auto pattn = [&](const float* zin, float* zout, int bp, unsigned short* wtq,
                   unsigned short* wtp) {
    ln(zin, P(bp), P(bp + 1), bfzn, RP);
    gemm(bfzn, wtq, P(bp + 3), nullptr, nullptr, bigbf, RP, 384, 128, 0);
    attn_kernel<<<dim3(16, 4, 256), 32, 0, stream>>>(bigbf, nullptr, bfA, L);
    gemm(bfA, wtp, P(bp + 5), zin, zout, nullptr, RP, 128, 128, 0);
  };
  pattn(pairB, pairA, PR, wt_prqkv, wt_prproj);       // row attention

  trpair_kernel<<<(RP * 128) / 256, 256, 0, stream>>>(pairA, f32A);
  pattn(f32A, f32B, PC, wt_pcqkv, wt_pcproj);         // column attention (transposed)
  trpair_kernel<<<(RP * 128) / 256, 256, 0, stream>>>(f32B, pairB);

  // ===== pair FFN =====
  ln(pairB, P(PF), P(PF + 1), bfzn, RP);
  gemm(bfzn, wt_fw1, P(PF + 3), nullptr, nullptr, bigbf, RP, 512, 128, 1);  // gelu -> bf16
  gemm(bigbf, wt_fw2, P(PF + 5), pairB, outPair, nullptr, RP, 128, 512, 0); // final pair
}
